// MultiHeadAttLoc_19602230739942
// MI455X (gfx1250) — matrix operand order repaired and checked by reading, not compiled, not run on hardware
//
#include <hip/hip_runtime.h>

// ---------------- problem constants ----------------
#define H_ 4
#define B_ 32
#define T_ 1500
#define E_ 512
#define D_ 512
#define A_ 256
#define C_ 10
#define KS_ 100
#define KW_ 201
#define O_ 512

typedef __attribute__((ext_vector_type(16))) __bf16 v16bf;
typedef __attribute__((ext_vector_type(8)))  float  v8f;

union BfPack {
    uint4 u[2];
    v16bf v;
};

// ---------------------------------------------------------------------------
// Kernel 1: W_enc [H,E,A] f32  ->  W_encT [H,A,E] bf16  (B-operand for WMMA)
// ---------------------------------------------------------------------------
__global__ void k_wencT(const float* __restrict__ wenc, __bf16* __restrict__ out) {
    int idx = blockIdx.x * 256 + threadIdx.x;      // < H*A*E = 524288
    int h   = idx >> 17;                            // A*E = 131072
    int rem = idx & 131071;
    int a   = rem >> 9;                             // E = 512
    int e   = rem & 511;
    out[idx] = (__bf16)wenc[((h << 9) + e) * A_ + a];
}

// ---------------------------------------------------------------------------
// Kernel 2: decb[h,b,a] = b_enc[h,a] + sum_d dec_z[b,d] * W_dec[h,d,a]
// ---------------------------------------------------------------------------
__global__ void k_decb(const float* __restrict__ dz, const float* __restrict__ wdec,
                       const float* __restrict__ benc, float* __restrict__ decb) {
    int h = blockIdx.x >> 5, b = blockIdx.x & 31;
    int a = threadIdx.x;                            // A_ == blockDim.x == 256
    __shared__ float z[D_];
    for (int i = threadIdx.x; i < D_; i += 256) z[i] = dz[b * D_ + i];
    __syncthreads();
    float s = benc[h * A_ + a];
    for (int d = 0; d < D_; ++d) s += z[d] * wdec[((h * D_ + d) << 8) + a];
    decb[((h * B_ + b) << 8) + a] = s;
}

// ---------------------------------------------------------------------------
// Kernel 3: conv_out[h,b,c,t] = sum_k att_prev[h,b,t+k-KS] * conv_w[h,c,k]
// ---------------------------------------------------------------------------
__global__ void k_conv(const float* __restrict__ ap, const float* __restrict__ cw,
                       float* __restrict__ convo) {
    int h = blockIdx.x >> 5, b = blockIdx.x & 31;
    __shared__ float apL[T_];
    __shared__ float cwL[C_ * KW_];
    for (int i = threadIdx.x; i < T_; i += 256) apL[i] = ap[(h * B_ + b) * T_ + i];
    for (int i = threadIdx.x; i < C_ * KW_; i += 256) cwL[i] = cw[h * C_ * KW_ + i];
    __syncthreads();
    for (int t = threadIdx.x; t < T_; t += 256) {
        int k0 = (KS_ - t) > 0 ? (KS_ - t) : 0;
        int k1 = (T_ + KS_ - t) < KW_ ? (T_ + KS_ - t) : KW_;
        for (int c = 0; c < C_; ++c) {
            float s = 0.f;
            for (int k = k0; k < k1; ++k) s += apL[t + k - KS_] * cwL[c * KW_ + k];
            convo[((h * B_ + b) * C_ + c) * T_ + t] = s;
        }
    }
}

// ---------------------------------------------------------------------------
// Kernel 4 (main): fused  e[h,b,t] = gvec . tanh(enc@W_enc + decb + att_conv)
// Workgroup = (tchunk of 64 rows, b, h); 8 waves.
// Wave w: M-tile = (w&3) (16 t-rows), N-half = (w>>2) (128 of 256 a-cols).
// K = E = 512, streamed through LDS in chunks of 64 (bf16, on-the-fly convert).
// Inner loop is register double-buffered so LDS fragment latency overlaps WMMA.
// ---------------------------------------------------------------------------
#define APITCH 72     // 64 + 8 bf16, 144B = 9*16B (aligned, conflict-free)
#define BPITCH 72

__launch_bounds__(256)
__global__ void k_score(const float*  __restrict__ enc,    // [B,T,E] f32
                        const __bf16* __restrict__ wencT,  // [H,A,E] bf16
                        const float*  __restrict__ decb,   // [H,B,A]
                        const float*  __restrict__ convo,  // [H,B,C,T]
                        const float*  __restrict__ watt,   // [H,C,A]
                        const float*  __restrict__ gvec,   // [H,A]
                        float* __restrict__ e_out) {       // [H,B,T]
    const int tc = blockIdx.x;           // 0..23
    const int b  = blockIdx.y;
    const int h  = blockIdx.z;
    const int tbase = tc * 64;
    const int tid  = threadIdx.x;
    const int lane = tid & 31;
    const int wave = tid >> 5;
    const int l15  = lane & 15;
    const bool lo  = lane < 16;
    const int mtile = wave & 3;
    const int nhalf = wave >> 2;

    __shared__ __bf16 aLds[64 * APITCH];     //  9216 B  enc tile (M x Kchunk)
    __shared__ __bf16 bLds[256 * BPITCH];    // 36864 B  W_encT tile (N x Kchunk)
    __shared__ float  decbL[A_];
    __shared__ float  gvL[A_];
    __shared__ float  wattL[C_ * A_];
    __shared__ float  convL[C_ * 64];
    __shared__ float  eL[64];

    // small preloads (visible after first barrier in the K loop)
    decbL[tid] = decb[((h * B_ + b) << 8) + tid];
    gvL[tid]   = gvec[(h << 8) + tid];
    for (int i = tid; i < C_ * A_; i += 256) wattL[i] = watt[h * C_ * A_ + i];
    for (int i = tid; i < C_ * 64; i += 256) {
        int c = i >> 6, tl = i & 63, t = tbase + tl;
        convL[i] = (t < T_) ? convo[((h * B_ + b) * C_ + c) * T_ + t] : 0.f;
    }
    if (tid < 64) eL[tid] = 0.f;

    v8f acc[8] = {};

    const __bf16* arow = &aLds[(mtile * 16 + l15) * APITCH];

    // fragment loaders (LDS -> regs), layouts per CDNA5 ISA:
    //  A: lane m=l15; lo lanes K {kb+0..7, kb+16..23}; hi lanes +8
    //  B: lane n=16j+l15; lo lanes K {kb+0..7, kb+16..23}; hi lanes +8
    auto loadA = [&](int kb) {
        BfPack p;
        const int ea = kb + (lo ? 0 : 8);
        p.u[0] = *(const uint4*)(arow + ea);
        p.u[1] = *(const uint4*)(arow + ea + 16);
        return p;
    };
    auto loadB = [&](int j, int kb) {
        BfPack p;
        const int n = nhalf * 128 + (j << 4) + l15;
        const __bf16* brow = &bLds[n * BPITCH + kb + (lo ? 0 : 8)];
        p.u[0] = *(const uint4*)(brow);
        p.u[1] = *(const uint4*)(brow + 16);
        return p;
    };

    for (int kc = 0; kc < 8; ++kc) {              // 8 chunks of K=64
        __syncthreads();
        // ---- load A: 64 t-rows x 64 e-cols, f32 -> bf16 (+ prefetch kc+1) ----
        #pragma unroll
        for (int i = 0; i < 4; ++i) {
            int q = tid + (i << 8);               // 0..1023 float4 units
            int row = q >> 4;
            int col = (q & 15) << 2;
            int t = tbase + row;
            float4 f = make_float4(0.f, 0.f, 0.f, 0.f);
            if (t < T_) {
                const float* gp = enc + ((size_t)(b * T_ + t)) * E_ + kc * 64 + col;
                f = *(const float4*)gp;
                if (kc < 7) __builtin_prefetch(gp + 64, 0, 0);   // next K-chunk
            }
            __bf16* dst = &aLds[row * APITCH + col];
            dst[0] = (__bf16)f.x; dst[1] = (__bf16)f.y;
            dst[2] = (__bf16)f.z; dst[3] = (__bf16)f.w;
        }
        // ---- load B: 256 a-rows x 64 e-cols bf16 (+ prefetch kc+1) ----
        #pragma unroll
        for (int i = 0; i < 8; ++i) {
            int q = tid + (i << 8);               // 0..2047 uint4 units
            int row = q >> 3;
            int off = (q & 7) << 3;               // bf16 offset, 8 per uint4
            const __bf16* gp = wencT + ((size_t)(h * A_ + row)) * E_ + kc * 64 + off;
            uint4 v = *(const uint4*)gp;
            if (kc < 7) __builtin_prefetch(gp + 64, 0, 0);       // next K-chunk
            *(uint4*)(&bLds[row * BPITCH + off]) = v;
        }
        __syncthreads();

        // ---- WMMA: 16 MACs (2 K-steps x 8 N-tiles), software pipelined ----
        BfPack ap_cur = loadA(0);
        BfPack bp_cur = loadB(0, 0);
        #pragma unroll
        for (int it = 0; it < 16; ++it) {
            const int j = it & 7;
            BfPack ap_nxt = ap_cur;
            BfPack bp_nxt = bp_cur;
            if (it + 1 < 16) {                    // prefetch next fragments
                const int itn = it + 1;
                const int kbn = (itn >> 3) << 5;
                if ((itn & 7) == 0) ap_nxt = loadA(kbn);
                bp_nxt = loadB(itn & 7, kbn);
            }
            acc[j] = __builtin_amdgcn_wmma_f32_16x16x32_bf16(
                false, ap_cur.v, false, bp_cur.v, (short)0, acc[j], false, false);
            ap_cur = ap_nxt;
            bp_cur = bp_nxt;
        }
    }

    // ---- epilogue: + decb + att_conv (K=C=10 VALU), tanh, dot gvec ----
    // acc[j] VGPR r: lanes 0-15 -> (t = base+r, a = 128*nhalf+16j+l15),
    //               lanes 16-31 -> (t = base+r+8)
    const int tl0 = mtile * 16 + (lo ? 0 : 8);
    float p[8];
    #pragma unroll
    for (int r = 0; r < 8; ++r) {
        const int tl = tl0 + r;
        float s = 0.f;
        #pragma unroll
        for (int j = 0; j < 8; ++j) {
            const int a = nhalf * 128 + (j << 4) + l15;
            float x = acc[j][r] + decbL[a];
            #pragma unroll
            for (int c = 0; c < C_; ++c)
                x += convL[(c << 6) + tl] * wattL[(c << 8) + a];
            s += tanhf(x) * gvL[a];
        }
        p[r] = s;
    }
    // reduce the 16 lanes of each half (xor net keeps halves separate)
    #pragma unroll
    for (int m = 1; m < 16; m <<= 1) {
        #pragma unroll
        for (int r = 0; r < 8; ++r) p[r] += __shfl_xor(p[r], m, 32);
    }
    if (l15 == 0) {
        #pragma unroll
        for (int r = 0; r < 8; ++r) atomicAdd(&eL[tl0 + r], p[r]);  // 2 N-halves combine
    }
    __syncthreads();
    if (tid < 64) {
        int t = tbase + tid;
        if (t < T_) e_out[(h * B_ + b) * T_ + t] = eL[tid];
    }
}

// ---------------------------------------------------------------------------
// Kernel 5: softmax over T with SCALING=2
// ---------------------------------------------------------------------------
__global__ void k_softmax(const float* __restrict__ e, float* __restrict__ w) {
    int hb = blockIdx.x, tid = threadIdx.x;
    const float* er = e + hb * T_;
    __shared__ float red[256];
    float m = -3.4e38f;
    for (int t = tid; t < T_; t += 256) m = fmaxf(m, er[t]);
    red[tid] = m; __syncthreads();
    for (int s2 = 128; s2 > 0; s2 >>= 1) {
        if (tid < s2) red[tid] = fmaxf(red[tid], red[tid + s2]);
        __syncthreads();
    }
    float mx = red[0]; __syncthreads();
    float s = 0.f;
    for (int t = tid; t < T_; t += 256) s += __expf(2.f * (er[t] - mx));
    red[tid] = s; __syncthreads();
    for (int s2 = 128; s2 > 0; s2 >>= 1) {
        if (tid < s2) red[tid] += red[tid + s2];
        __syncthreads();
    }
    float inv = 1.f / red[0];
    for (int t = tid; t < T_; t += 256) w[hb * T_ + t] = __expf(2.f * (er[t] - mx)) * inv;
}

// ---------------------------------------------------------------------------
// Kernel 6: c_cat[b, h*E+e] = sum_t w[h,b,t] * enc[b,t,e]   (reads enc once)
// ---------------------------------------------------------------------------
__global__ void k_context(const float* __restrict__ w, const float* __restrict__ enc,
                          float* __restrict__ c_cat) {
    int b = blockIdx.y;
    int ecol = blockIdx.x * 256 + threadIdx.x;     // 0..511
    __shared__ float wl[H_ * T_];                   // 24 KB
    for (int i = threadIdx.x; i < H_ * T_; i += 256) {
        int h = i / T_, t = i % T_;
        wl[i] = w[(h * B_ + b) * T_ + t];
    }
    __syncthreads();
    float acc0 = 0.f, acc1 = 0.f, acc2 = 0.f, acc3 = 0.f;
    const float* ep = enc + (size_t)b * T_ * E_ + ecol;
    for (int t = 0; t < T_; ++t) {
        float v = ep[(size_t)t * E_];
        acc0 += wl[t] * v;
        acc1 += wl[T_ + t] * v;
        acc2 += wl[2 * T_ + t] * v;
        acc3 += wl[3 * T_ + t] * v;
    }
    c_cat[b * (H_ * E_) + 0 * E_ + ecol] = acc0;
    c_cat[b * (H_ * E_) + 1 * E_ + ecol] = acc1;
    c_cat[b * (H_ * E_) + 2 * E_ + ecol] = acc2;
    c_cat[b * (H_ * E_) + 3 * E_ + ecol] = acc3;
}

// ---------------------------------------------------------------------------
// Kernel 7: c[b,o] = b_o[o] + c_cat[b,:] @ W_o[:,o]
// ---------------------------------------------------------------------------
__global__ void k_outproj(const float* __restrict__ c_cat, const float* __restrict__ Wo,
                          const float* __restrict__ bo, float* __restrict__ out_c) {
    int b = blockIdx.y;
    int o = blockIdx.x * 256 + threadIdx.x;        // 0..511
    __shared__ float cl[H_ * E_];                   // 8 KB
    for (int i = threadIdx.x; i < H_ * E_; i += 256) cl[i] = c_cat[b * (H_ * E_) + i];
    __syncthreads();
    float s = bo[o];
    for (int k = 0; k < H_ * E_; ++k) s += cl[k] * Wo[k * O_ + o];
    out_c[b * O_ + o] = s;
}

// ---------------------------------------------------------------------------
extern "C" void kernel_launch(void* const* d_in, const int* in_sizes, int n_in,
                              void* d_out, int out_size, void* d_ws, size_t ws_size,
                              hipStream_t stream) {
    const float* enc      = (const float*)d_in[0];
    /* d_in[1] = enc_len (unused by the reference computation) */
    const float* dec_z    = (const float*)d_in[2];
    const float* att_prev = (const float*)d_in[3];
    const float* Wenc     = (const float*)d_in[4];
    const float* benc     = (const float*)d_in[5];
    const float* Wdec     = (const float*)d_in[6];
    const float* Watt     = (const float*)d_in[7];
    const float* convw    = (const float*)d_in[8];
    const float* gvec     = (const float*)d_in[9];
    const float* Wo       = (const float*)d_in[10];
    const float* bo       = (const float*)d_in[11];

    float* out_c = (float*)d_out;                  // [B,O]
    float* out_w = out_c + B_ * O_;                // [H,B,T]

    char* ws = (char*)d_ws;
    __bf16* wencT = (__bf16*)ws;  ws += (size_t)H_ * A_ * E_ * 2;        // 1.0 MB
    float*  decb  = (float*)ws;   ws += (size_t)H_ * B_ * A_ * 4;        // 128 KB
    float*  convo = (float*)ws;   ws += (size_t)H_ * B_ * C_ * T_ * 4;   // 7.7 MB
    float*  e_buf = (float*)ws;   ws += (size_t)H_ * B_ * T_ * 4;        // 768 KB
    float*  c_cat = (float*)ws;                                          // 256 KB

    k_wencT  <<<(H_ * A_ * E_) / 256, 256, 0, stream>>>(Wenc, wencT);
    k_decb   <<<H_ * B_, 256, 0, stream>>>(dec_z, Wdec, benc, decb);
    k_conv   <<<H_ * B_, 256, 0, stream>>>(att_prev, convw, convo);
    k_score  <<<dim3(24, B_, H_), 256, 0, stream>>>(enc, wencT, decb, convo, Watt, gvec, e_buf);
    k_softmax<<<H_ * B_, 256, 0, stream>>>(e_buf, out_w);
    k_context<<<dim3(2, B_), 256, 0, stream>>>(out_w, enc, c_cat);
    k_outproj<<<dim3(2, B_), 256, 0, stream>>>(c_cat, Wo, bo, out_c);
}
